// MoELayer_1606317769176
// MI455X (gfx1250) — matrix operand order repaired and checked by reading, not compiled, not run on hardware
//
#include <hip/hip_runtime.h>

#define DIM   512
#define DFF   2048
#define NEXP  16
#define TTOK  16384     // B*S tokens
#define TILE_M 128
#define CHUNK  32       // DFF chunk per pipeline stage
#define NCHUNK (DFF / CHUNK)
#define FFN_THREADS 512

// LDS layout (bf16 elements), rows padded for bank spread + 16B alignment
#define US   520  // U tile row stride      [128][520]
#define W1S  520  // W1^T chunk row stride  [32][520]  (n-major, k contiguous)
#define W2S  40   // W2^T chunk row stride  [512][40]  (n-major, k contiguous)
#define HS   40   // H tile row stride      [128][40]
#define U_OFF    0
#define W1A_OFF  (TILE_M * US * 2)                 // 133120
#define W1B_OFF  (W1A_OFF + CHUNK * W1S * 2)       // 166400
#define W2A_OFF  (W1B_OFF + CHUNK * W1S * 2)       // 199680
#define W2B_OFF  (W2A_OFF + DIM * W2S * 2)         // 240640
#define H_OFF    (W2B_OFF + DIM * W2S * 2)         // 281600
#define SMEM_BYTES (H_OFF + TILE_M * HS * 2)       // 291840

typedef __attribute__((ext_vector_type(16))) __bf16 v16bf;
typedef __attribute__((ext_vector_type(8)))  __bf16 v8bf;
typedef __attribute__((ext_vector_type(4)))  __bf16 v4bf;
typedef __attribute__((ext_vector_type(8)))  float  v8f;
typedef __attribute__((ext_vector_type(4)))  float  v4f;

static __device__ __forceinline__ v16bf cat8(v8bf lo, v8bf hi) {
    return __builtin_shufflevector(lo, hi, 0,1,2,3,4,5,6,7,8,9,10,11,12,13,14,15);
}

// A operand (16x32 bf16, row-major source): lane holds row (lane&15),
// K = kb..kb+7 and kb+16..kb+23 with kb = 8*(lane>=16).
static __device__ __forceinline__ v16bf load_frag_a(const __bf16* base, int row0,
                                                    int rs, int k0, int lane) {
    int row = row0 + (lane & 15);
    int kb  = (lane >> 4) << 3;
    const __bf16* p = base + row * rs + k0 + kb;
    v8bf lo = *(const v8bf*)p;
    v8bf hi = *(const v8bf*)(p + 16);
    return cat8(lo, hi);
}

// B operand (32x16 bf16) from a transposed [N][K] LDS image:
// lane holds col (lane&15), K = kb..kb+7 and kb+16..kb+23 with kb = 8*(lane>=16).
static __device__ __forceinline__ v16bf load_frag_b(const __bf16* baseT, int col0,
                                                    int rs, int k0, int lane) {
    int col = col0 + (lane & 15);
    int kb  = (lane >> 4) << 3;
    const __bf16* p = baseT + col * rs + k0 + kb;
    v8bf lo = *(const v8bf*)p;
    v8bf hi = *(const v8bf*)(p + 16);
    return cat8(lo, hi);
}

static __device__ __forceinline__ v8f wmma_bf16(v16bf a, v16bf b, v8f c) {
    return __builtin_amdgcn_wmma_f32_16x16x32_bf16(false, a, false, b,
                                                   (short)0, c, false, false);
}

// async global->LDS 16B/lane copy (CDNA5, ASYNCcnt-tracked)
static __device__ __forceinline__ void async_copy16(unsigned lds_byte,
                                                    unsigned long long gaddr) {
    asm volatile("global_load_async_to_lds_b128 %0, %1, off"
                 :: "v"(lds_byte), "v"(gaddr) : "memory");
}
static __device__ __forceinline__ void wait_async0() {
    asm volatile("s_wait_asynccnt 0" ::: "memory");
}

// ---------------------------------------------------------------- kernel 0
__global__ void moe_zero_kernel(float* __restrict__ out, int* __restrict__ counts,
                                size_t n4) {
    if (blockIdx.x == 0 && threadIdx.x < NEXP) counts[threadIdx.x] = 0;
    size_t stride = (size_t)gridDim.x * blockDim.x;
    v4f z = {0.f, 0.f, 0.f, 0.f};
    for (size_t i = (size_t)blockIdx.x * blockDim.x + threadIdx.x; i < n4; i += stride)
        ((v4f*)out)[i] = z;
}

// ---------------------------------------------------------------- kernel P
// fp32 [R][C] -> bf16 [C][R] per expert (LDS-tiled transpose + convert)
__global__ void moe_transpose_cvt(const float* __restrict__ src,
                                  __bf16* __restrict__ dst, int R, int C) {
    __shared__ float tile[32][33];
    const int e = blockIdx.z;
    const float* s = src + (size_t)e * R * C;
    __bf16*      d = dst + (size_t)e * R * C;
    int x = blockIdx.x * 32 + threadIdx.x;          // col in src
#pragma unroll
    for (int j = 0; j < 4; ++j) {
        int y = blockIdx.y * 32 + threadIdx.y + j * 8;  // row in src
        tile[threadIdx.y + j * 8][threadIdx.x] = s[(size_t)y * C + x];
    }
    __syncthreads();
    int x2 = blockIdx.y * 32 + threadIdx.x;         // col in dst (= row in src)
#pragma unroll
    for (int j = 0; j < 4; ++j) {
        int y2 = blockIdx.x * 32 + threadIdx.y + j * 8; // row in dst (= col in src)
        d[(size_t)y2 * R + x2] = (__bf16)tile[threadIdx.x][threadIdx.y + j * 8];
    }
}

// ---------------------------------------------------------------- kernel 1
// One wave per token: scores = u @ centroids^T, softmax, top-2, append to
// per-expert gather lists.
__global__ void moe_gate_kernel(const float* __restrict__ u,
                                const float* __restrict__ cent,
                                int* __restrict__ counts,
                                int* __restrict__ lists,
                                float* __restrict__ gates) {
    __shared__ float c_lds[NEXP * DIM];
    for (int i = threadIdx.x; i < NEXP * DIM; i += blockDim.x) c_lds[i] = cent[i];
    __syncthreads();

    const int lane = threadIdx.x & 31;
    const int wave = threadIdx.x >> 5;
    const int tok  = blockIdx.x * (blockDim.x >> 5) + wave;
    if (tok >= TTOK) return;

    float acc[NEXP];
#pragma unroll
    for (int e = 0; e < NEXP; ++e) acc[e] = 0.f;

    const float* urow = u + (size_t)tok * DIM;
#pragma unroll
    for (int it = 0; it < 4; ++it) {
        int c = it * 128 + lane * 4;
        v4f uv = *(const v4f*)(urow + c);
#pragma unroll
        for (int e = 0; e < NEXP; ++e) {
            v4f cv = *(const v4f*)(c_lds + e * DIM + c);
            acc[e] += uv[0]*cv[0] + uv[1]*cv[1] + uv[2]*cv[2] + uv[3]*cv[3];
        }
    }
#pragma unroll
    for (int e = 0; e < NEXP; ++e) {
        float v = acc[e];
#pragma unroll
        for (int off = 16; off > 0; off >>= 1) v += __shfl_xor(v, off, 32);
        acc[e] = v;
    }

    if (lane == 0) {
        float m1 = -1e30f; int i1 = 0;
#pragma unroll
        for (int e = 0; e < NEXP; ++e) if (acc[e] > m1) { m1 = acc[e]; i1 = e; }
        float m2 = -1e30f; int i2 = (i1 == 0) ? 1 : 0;
#pragma unroll
        for (int e = 0; e < NEXP; ++e)
            if (e != i1 && acc[e] > m2) { m2 = acc[e]; i2 = e; }
        float den = 0.f;
#pragma unroll
        for (int e = 0; e < NEXP; ++e) den += __expf(acc[e] - m1);
        float g1 = 1.0f / den;
        float g2 = __expf(m2 - m1) / den;

        int p1 = atomicAdd(&counts[i1], 1);
        lists[i1 * TTOK + p1] = tok;  gates[i1 * TTOK + p1] = g1;
        int p2 = atomicAdd(&counts[i2], 1);
        lists[i2 * TTOK + p2] = tok;  gates[i2 * TTOK + p2] = g2;
    }
}

// ---------------------------------------------------------------- kernel 2
// Fused expert FFN with async double-buffered weight staging:
// out[tok] += gate * (relu(U@W1 + b1) @ W2 + b2)
__global__ void __launch_bounds__(FFN_THREADS, 1)
moe_ffn_kernel(const float* __restrict__ u,
               const __bf16* __restrict__ W1T,   // [NEXP][DFF][DIM] bf16
               const float*  __restrict__ b1,
               const __bf16* __restrict__ W2T,   // [NEXP][DIM][DFF] bf16
               const float*  __restrict__ b2,
               const int* __restrict__ counts, const int* __restrict__ lists,
               const float* __restrict__ gates, float* __restrict__ out) {
    extern __shared__ __align__(16) char smem[];
    __bf16* Ub = (__bf16*)(smem + U_OFF);     // [128][US]
    __bf16* Hb = (__bf16*)(smem + H_OFF);     // [128][HS]

    const int e    = blockIdx.y;
    const int tile = blockIdx.x;
    const int cnt  = counts[e];
    if (tile * TILE_M >= cnt) return;

    const int t    = threadIdx.x;
    const int lane = t & 31;
    const int w    = t >> 5;

    const int*   list = lists + e * TTOK + tile * TILE_M;
    const float* gate = gates + e * TTOK + tile * TILE_M;

    const unsigned lds0 = (unsigned)(uintptr_t)(void*)smem;  // LDS byte base
    const unsigned long long w1g = (unsigned long long)(uintptr_t)(W1T + (size_t)e * DFF * DIM);
    const unsigned long long w2g = (unsigned long long)(uintptr_t)(W2T + (size_t)e * DIM * DFF);

    // issue async staging of chunk ci into buffer (ci&1)
    auto issue_chunk = [&](int ci) {
        const int kk = ci * CHUNK;
        const unsigned w1dst = lds0 + ((ci & 1) ? W1B_OFF : W1A_OFF);
        const unsigned w2dst = lds0 + ((ci & 1) ? W2B_OFF : W2A_OFF);
        // W1T chunk: rows n=kk..kk+31, each 512 bf16 contiguous (64 segs/row)
#pragma unroll
        for (int it = 0; it < 4; ++it) {
            int idx = it * FFN_THREADS + t;        // 0..2047
            int row = idx >> 6, seg = idx & 63;
            async_copy16(w1dst + (unsigned)(row * (W1S * 2) + seg * 16),
                         w1g + (unsigned long long)(kk + row) * (DIM * 2) + seg * 16);
        }
        // W2T chunk: 512 rows n, 32 bf16 each (4 segs/row), global row stride DFF
#pragma unroll
        for (int it = 0; it < 4; ++it) {
            int idx = it * FFN_THREADS + t;        // 0..2047
            int n = idx >> 2, seg = idx & 3;
            async_copy16(w2dst + (unsigned)(n * (W2S * 2) + seg * 16),
                         w2g + ((unsigned long long)n * DFF + kk) * 2 + seg * 16);
        }
    };

    // ---- gather + convert U tile: wave w loads rows w*8..w*8+7 ----
    for (int j = 0; j < 8; ++j) {
        int  r     = w * 8 + j;
        bool valid = (tile * TILE_M + r) < cnt;
        int  tokr  = valid ? list[r] : 0;
        const float* urow = u + (size_t)tokr * DIM;
#pragma unroll
        for (int it = 0; it < 4; ++it) {
            int c = it * 128 + lane * 4;
            v4f uv = {0.f, 0.f, 0.f, 0.f};
            if (valid) uv = *(const v4f*)(urow + c);
            v4bf bv;
            bv[0] = (__bf16)uv[0]; bv[1] = (__bf16)uv[1];
            bv[2] = (__bf16)uv[2]; bv[3] = (__bf16)uv[3];
            *(v4bf*)(Ub + r * US + c) = bv;
        }
    }

    const int hm = w & 7, hn = w >> 3;          // stage-A tile (8 x 2 tiles)
    const int wm = w & 3, wn = w >> 2;          // stage-B tile assignment

    v8f C[2][8];
#pragma unroll
    for (int mt = 0; mt < 2; ++mt)
#pragma unroll
        for (int nt = 0; nt < 8; ++nt)
#pragma unroll
            for (int v = 0; v < 8; ++v) C[mt][nt][v] = 0.f;

    issue_chunk(0);
    wait_async0();
    __syncthreads();   // U + chunk0 resident

    for (int ci = 0; ci < NCHUNK; ++ci) {
        const int kk = ci * CHUNK;
        const __bf16* W1c = (const __bf16*)(smem + ((ci & 1) ? W1B_OFF : W1A_OFF));
        const __bf16* W2c = (const __bf16*)(smem + ((ci & 1) ? W2B_OFF : W2A_OFF));

        if (ci + 1 < NCHUNK) issue_chunk(ci + 1);   // overlap DMA with compute

        // ---- stage A: H[128x32] = relu(U @ W1chunk + b1) ----
        v8f hA = {};
        for (int ks = 0; ks < DIM; ks += 32) {
            v16bf a = load_frag_a(Ub, hm * 16, US, ks, lane);
            v16bf b = load_frag_b(W1c, hn * 16, W1S, ks, lane);
            hA = wmma_bf16(a, b, hA);
        }
        {
            int ncol  = lane & 15;
            int rbase = hm * 16 + ((lane >> 4) << 3);
            float bias = b1[(size_t)e * DFF + kk + hn * 16 + ncol];
#pragma unroll
            for (int v = 0; v < 8; ++v) {
                float h = hA[v] + bias; h = h > 0.f ? h : 0.f;
                Hb[(rbase + v) * HS + hn * 16 + ncol] = (__bf16)h;
            }
        }
        __syncthreads();   // H visible

        // ---- stage B: C[32x128 per wave] += H @ W2chunk ----
        {
            v16bf a0 = load_frag_a(Hb, wm * 32,      HS, 0, lane);
            v16bf a1 = load_frag_a(Hb, wm * 32 + 16, HS, 0, lane);
#pragma unroll
            for (int nt = 0; nt < 8; ++nt) {
                v16bf b = load_frag_b(W2c, wn * 128 + nt * 16, W2S, 0, lane);
                C[0][nt] = wmma_bf16(a0, b, C[0][nt]);
                C[1][nt] = wmma_bf16(a1, b, C[1][nt]);
            }
        }

        wait_async0();     // next chunk has landed (this wave's transfers)
        __syncthreads();   // all waves done with this buffer + next chunk ready
    }

    // ---- epilogue: out[tok] += gate * (C + b2) ----
    {
        int ncol = lane & 15;
        int radd = (lane >> 4) << 3;
#pragma unroll
        for (int mt = 0; mt < 2; ++mt) {
#pragma unroll
            for (int v = 0; v < 8; ++v) {
                int m   = wm * 32 + mt * 16 + radd + v;
                int row = tile * TILE_M + m;
                if (row < cnt) {
                    int   tok = list[m];
                    float g   = gate[m];
#pragma unroll
                    for (int nt = 0; nt < 8; ++nt) {
                        int colg = wn * 128 + nt * 16 + ncol;
                        float val = g * (C[mt][nt][v] + b2[(size_t)e * DIM + colg]);
                        unsafeAtomicAdd(out + (size_t)tok * DIM + colg, val);
                    }
                }
            }
        }
    }
}

// ---------------------------------------------------------------- launch
extern "C" void kernel_launch(void* const* d_in, const int* in_sizes, int n_in,
                              void* d_out, int out_size, void* d_ws, size_t ws_size,
                              hipStream_t stream) {
    const float* u    = (const float*)d_in[0];
    const float* cent = (const float*)d_in[1];
    const float* W1   = (const float*)d_in[2];
    const float* b1   = (const float*)d_in[3];
    const float* W2   = (const float*)d_in[4];
    const float* b2   = (const float*)d_in[5];
    float*       out  = (float*)d_out;

    // workspace layout (256B-aligned sections)
    char* ws = (char*)d_ws;
    int*    counts = (int*)ws;                                      // 16 ints
    int*    lists  = (int*)(ws + 256);                              // NEXP*T ints
    float*  gates  = (float*)(ws + 256 + (size_t)NEXP * TTOK * 4);  // NEXP*T floats
    __bf16* W1Tg   = (__bf16*)(ws + 2097408);                       // 33.5 MB
    __bf16* W2Tg   = (__bf16*)(ws + 2097408 + (size_t)NEXP * DIM * DFF * 2);

    moe_zero_kernel<<<4096, 256, 0, stream>>>(out, counts, (size_t)TTOK * DIM / 4);

    // weights: fp32 -> bf16, transposed so K is contiguous per output column
    moe_transpose_cvt<<<dim3(DFF / 32, DIM / 32, NEXP), dim3(32, 8), 0, stream>>>(
        W1, W1Tg, DIM, DFF);
    moe_transpose_cvt<<<dim3(DIM / 32, DFF / 32, NEXP), dim3(32, 8), 0, stream>>>(
        W2, W2Tg, DFF, DIM);

    moe_gate_kernel<<<TTOK / 8, 256, 0, stream>>>(u, cent, counts, lists, gates);

    hipFuncSetAttribute((const void*)moe_ffn_kernel,
                        hipFuncAttributeMaxDynamicSharedMemorySize, SMEM_BYTES);
    dim3 grid(TTOK / TILE_M, NEXP);
    moe_ffn_kernel<<<grid, FFN_THREADS, SMEM_BYTES, stream>>>(
        u, W1Tg, b1, W2Tg, b2, counts, lists, gates, out);
}
